// VIMModule_9156870275463
// MI455X (gfx1250) — matrix operand order repaired and checked by reading, not compiled, not run on hardware
//
#include <hip/hip_runtime.h>
#include <hip/hip_bf16.h>

// ---------------------------------------------------------------------------
// Mamba block forward for MI455X (gfx1250, wave32, WMMA).
// GEMMs use v_wmma_f32_16x16x32_f16 (f16 inputs, f32 accumulate); the
// sequential selective scan runs in fp32 with per-channel state in registers.
// All GEMMs are branch-free in the K-loop (N padded to a multiple of 64).
// ---------------------------------------------------------------------------

#define Bq   4
#define Lq   2048
#define Dq   1024
#define DINq 2048          // 2*D
#define Nq   16
#define DTRq 64            // D/16
#define BLq  (Bq * Lq)     // 8192 rows for all GEMMs
#define XDS  128           // padded row stride of x_dbl (96 -> 128)

typedef _Float16 h8  __attribute__((ext_vector_type(8)));
typedef _Float16 h16 __attribute__((ext_vector_type(16)));
typedef float    f8  __attribute__((ext_vector_type(8)));

// ---------------------------------------------------------------------------
// C = A * B^T WMMA GEMM, branch-free inner loop.
//   A: M x K row-major f16,  B: N x K row-major f16,  C: M x N f32.
// Wave tile 32x64 (2 A-frags x 4 B-frags = 8 WMMA / 32-K step).
// Block = 8 waves -> 256x64 of C.  Requires M%256==0, N%64==0, K%32==0.
// ---------------------------------------------------------------------------
__global__ __launch_bounds__(256) void gemm_wmma_f16(
    const _Float16* __restrict__ A, const _Float16* __restrict__ B,
    float* __restrict__ C, int M, int N, int K)
{
    const int lane = threadIdx.x & 31;   // wave32
    const int wave = threadIdx.x >> 5;
    const int r    = lane & 15;
    const int kh   = lane >> 4;
    const int m0   = blockIdx.y * 256 + wave * 32;
    const int n0   = blockIdx.x * 64;

    f8 acc0[4] = {f8{}, f8{}, f8{}, f8{}};
    f8 acc1[4] = {f8{}, f8{}, f8{}, f8{}};

    const _Float16* __restrict__ Arow0 = A + (size_t)(m0 + r) * K;
    const _Float16* __restrict__ Arow1 = A + (size_t)(m0 + 16 + r) * K;

    for (int k0 = 0; k0 < K; k0 += 32) {
        // Speculative prefetch of the next K chunk (dropped if OOB).
        __builtin_prefetch(Arow0 + k0 + 32, 0, 0);   // global_prefetch_b8
        __builtin_prefetch(Arow1 + k0 + 32, 0, 0);

        // A fragments: lane (r, kh) holds K = kh*8..+8 and 16+kh*8..+8 of row r.
        h8 a0lo = *(const h8*)(Arow0 + k0 + kh * 8);
        h8 a0hi = *(const h8*)(Arow0 + k0 + 16 + kh * 8);
        h16 a0 = __builtin_shufflevector(a0lo, a0hi,
            0,1,2,3,4,5,6,7,8,9,10,11,12,13,14,15);
        h8 a1lo = *(const h8*)(Arow1 + k0 + kh * 8);
        h8 a1hi = *(const h8*)(Arow1 + k0 + 16 + kh * 8);
        h16 a1 = __builtin_shufflevector(a1lo, a1hi,
            0,1,2,3,4,5,6,7,8,9,10,11,12,13,14,15);

        #pragma unroll
        for (int j = 0; j < 4; ++j) {
            // B fragment: lane (r, kh) holds K = kh*8..+8 and 16+kh*8..+8 of row n.
            const _Float16* __restrict__ Brow =
                B + (size_t)(n0 + j * 16 + r) * K + k0 + kh * 8;
            h8 blo = *(const h8*)(Brow);
            h8 bhi = *(const h8*)(Brow + 16);
            h16 bf = __builtin_shufflevector(blo, bhi,
                0,1,2,3,4,5,6,7,8,9,10,11,12,13,14,15);
            acc0[j] = __builtin_amdgcn_wmma_f32_16x16x32_f16(
                false, a0, false, bf, (short)0, acc0[j], false, false);
            acc1[j] = __builtin_amdgcn_wmma_f32_16x16x32_f16(
                false, a1, false, bf, (short)0, acc1[j], false, false);
        }
    }

    // C/D layout: VGPR v -> row kh*8+v, lane r -> column (coalesced stores).
    #pragma unroll
    for (int j = 0; j < 4; ++j) {
        const int col = n0 + j * 16 + r;
        #pragma unroll
        for (int v = 0; v < 8; ++v) {
            C[(size_t)(m0 + kh * 8 + v) * N + col]      = acc0[j][v];
            C[(size_t)(m0 + 16 + kh * 8 + v) * N + col] = acc1[j][v];
        }
    }
}

// ---------------------------------------------------------------------------
// resid = residual + hidden ; h16 = LayerNorm(resid) converted to f16.
// One block per row of D=1024, 256 threads x 4 elements.
// ---------------------------------------------------------------------------
__global__ __launch_bounds__(256) void add_ln_kernel(
    const float* __restrict__ hs, const float* __restrict__ res,
    const float* __restrict__ w, const float* __restrict__ bia,
    float* __restrict__ resid_out, _Float16* __restrict__ h16out)
{
    __shared__ float s1[256], s2[256];
    const size_t row = blockIdx.x;
    const float* __restrict__ hrow = hs + row * Dq;
    const float* __restrict__ rrow = res + row * Dq;

    float v[4], sum = 0.f, sq = 0.f;
    #pragma unroll
    for (int j = 0; j < 4; ++j) {
        int c = threadIdx.x + j * 256;
        float t = hrow[c] + rrow[c];
        v[j] = t; sum += t; sq += t * t;
    }
    s1[threadIdx.x] = sum; s2[threadIdx.x] = sq;
    __syncthreads();
    for (int off = 128; off > 0; off >>= 1) {
        if ((int)threadIdx.x < off) {
            s1[threadIdx.x] += s1[threadIdx.x + off];
            s2[threadIdx.x] += s2[threadIdx.x + off];
        }
        __syncthreads();
    }
    const float mu  = s1[0] * (1.0f / Dq);
    const float var = s2[0] * (1.0f / Dq) - mu * mu;
    const float rs  = rsqrtf(var + 1e-5f);

    float* __restrict__ ro = resid_out + row * Dq;
    _Float16* __restrict__ ho = h16out + row * Dq;
    #pragma unroll
    for (int j = 0; j < 4; ++j) {
        int c = threadIdx.x + j * 256;
        ro[c] = v[j];
        ho[c] = (_Float16)((v[j] - mu) * rs * w[c] + bia[c]);
    }
}

// ---------------------------------------------------------------------------
// Depthwise causal conv (DCONV=4) over time + SiLU. Reads the x half of xz
// ((BL,4096) row-major), writes f32 (for the scan) and f16 (for x_proj GEMM).
// ---------------------------------------------------------------------------
__global__ __launch_bounds__(256) void conv_silu_kernel(
    const float* __restrict__ xz, const float* __restrict__ cw,
    const float* __restrict__ cb, float* __restrict__ xc32,
    _Float16* __restrict__ xc16)
{
    const size_t idx = (size_t)blockIdx.x * 256 + threadIdx.x; // < BL*DIN
    const int d  = (int)(idx & (DINq - 1));
    const size_t bl = idx >> 11;          // b*L + l   (DIN = 2048 = 2^11)
    const int l  = (int)(bl & (Lq - 1));

    float acc = cb[d];
    #pragma unroll
    for (int j = 0; j < 4; ++j) {
        const int ll = l - 3 + j;
        const float vv = (ll >= 0) ? xz[(bl - 3 + j) * (2 * DINq) + d] : 0.0f;
        acc += cw[d * 4 + j] * vv;
    }
    const float s = acc / (1.0f + __expf(-acc));   // SiLU
    xc32[idx] = s;
    xc16[idx] = (_Float16)s;
}

// Extract dt_raw (first 64 cols of x_dbl, row stride XDS) as f16.
__global__ __launch_bounds__(256) void extract_dt_kernel(
    const float* __restrict__ xdbl, _Float16* __restrict__ dtr16)
{
    const int i = blockIdx.x * 256 + threadIdx.x;     // < BL*DTR
    const int row = i >> 6, c = i & 63;
    dtr16[i] = (_Float16)xdbl[(size_t)row * XDS + c];
}

// dt = softplus(dt_proj_out + dt_bias), in place.
__global__ __launch_bounds__(256) void softplus_kernel(
    float* __restrict__ dt, const float* __restrict__ bias)
{
    const size_t i = (size_t)blockIdx.x * 256 + threadIdx.x;
    const float v = dt[i] + bias[i & (DINq - 1)];
    dt[i] = (v > 20.0f) ? v : log1pf(__expf(v));
}

// ---------------------------------------------------------------------------
// Selective scan: one thread per (b, d) channel; N=16 state in registers;
// sequential loop over L=2048. Fuses y = (scan + x*D) * silu(z) and emits
// y as f16 for the out_proj GEMM plus the final state.
// ---------------------------------------------------------------------------
__global__ __launch_bounds__(256) void scan_kernel(
    const float* __restrict__ dt32, const float* __restrict__ xc32,
    const float* __restrict__ xz32, const float* __restrict__ xdbl,
    const float* __restrict__ A_log, const float* __restrict__ Dp,
    _Float16* __restrict__ y16, float* __restrict__ laststate)
{
    const int b = blockIdx.x >> 3;                          // 8 blocks per b
    const int d = ((blockIdx.x & 7) << 8) + threadIdx.x;    // channel

    float Arow[Nq], st[Nq];
    #pragma unroll
    for (int n = 0; n < Nq; ++n) {
        Arow[n] = -__expf(A_log[d * Nq + n]);
        st[n] = 0.0f;
    }
    const float dval = Dp[d];

    for (int l = 0; l < Lq; ++l) {
        const size_t row = (size_t)b * Lq + l;
        const float dt = dt32[row * DINq + d];
        const float xv = xc32[row * DINq + d];
        const float zv = xz32[row * (2 * DINq) + DINq + d];
        const float dx = dt * xv;
        const float* __restrict__ bc = xdbl + row * XDS;

        float y = 0.0f;
        #pragma unroll
        for (int n = 0; n < Nq; ++n) {
            const float Bn = bc[DTRq + n];        // wave-uniform address
            const float Cn = bc[DTRq + Nq + n];
            st[n] = __expf(dt * Arow[n]) * st[n] + dx * Bn;
            y += st[n] * Cn;
        }
        const float yf = (y + xv * dval) * (zv / (1.0f + __expf(-zv)));
        y16[row * DINq + d] = (_Float16)yf;
    }

    #pragma unroll
    for (int n = 0; n < Nq; ++n)
        laststate[((size_t)b * DINq + d) * Nq + n] = st[n];
}

// f32 -> f16 conversion; elements >= n_valid are written as zero (padding).
__global__ __launch_bounds__(256) void f32_to_f16_kernel(
    const float* __restrict__ in, _Float16* __restrict__ out,
    int n_valid, int n_total)
{
    const int i = blockIdx.x * 256 + threadIdx.x;
    if (i < n_total) out[i] = (i < n_valid) ? (_Float16)in[i] : (_Float16)0.0f;
}

// ---------------------------------------------------------------------------
extern "C" void kernel_launch(void* const* d_in, const int* in_sizes, int n_in,
                              void* d_out, int out_size, void* d_ws, size_t ws_size,
                              hipStream_t stream) {
    (void)in_sizes; (void)n_in; (void)out_size; (void)ws_size;

    const float* hidden     = (const float*)d_in[0];
    const float* residual   = (const float*)d_in[1];
    const float* norm_w     = (const float*)d_in[2];
    const float* norm_b     = (const float*)d_in[3];
    const float* in_proj_w  = (const float*)d_in[4];   // (2*DIN, D)
    const float* conv_w     = (const float*)d_in[5];   // (DIN, 1, 4)
    const float* conv_b     = (const float*)d_in[6];
    const float* x_proj_w   = (const float*)d_in[7];   // (96, DIN)
    const float* dt_proj_w  = (const float*)d_in[8];   // (DIN, 64)
    const float* dt_proj_b  = (const float*)d_in[9];
    const float* A_log      = (const float*)d_in[10];  // (DIN, N)
    const float* D_param    = (const float*)d_in[11];
    const float* out_proj_w = (const float*)d_in[12];  // (D, DIN)
    // d_in[13] = layer (unused)

    // Output regions: y (B,L,D) | last_state (B,DIN,N) | resid (B,L,D)
    float* out_y     = (float*)d_out;
    float* out_state = out_y + (size_t)BLq * Dq;
    float* out_resid = out_state + (size_t)Bq * DINq * Nq;

    // Scratch carve-out (256B aligned). Total ~370 MB.
    char* wp = (char*)d_ws;
    auto carve = [&](size_t bytes) -> char* {
        char* p = wp;
        wp += (bytes + 255) & ~(size_t)255;
        return p;
    };
    _Float16* h16    = (_Float16*)carve((size_t)BLq * Dq * 2);
    _Float16* win16  = (_Float16*)carve((size_t)2 * DINq * Dq * 2);
    _Float16* wxp16  = (_Float16*)carve((size_t)XDS * DINq * 2);      // padded
    _Float16* wdt16  = (_Float16*)carve((size_t)DINq * DTRq * 2);
    _Float16* wout16 = (_Float16*)carve((size_t)Dq * DINq * 2);
    float*    xz32   = (float*)carve((size_t)BLq * 2 * DINq * 4);
    float*    xc32   = (float*)carve((size_t)BLq * DINq * 4);
    _Float16* xc16   = (_Float16*)carve((size_t)BLq * DINq * 2);
    float*    xdbl32 = (float*)carve((size_t)BLq * XDS * 4);          // padded
    _Float16* dtr16  = (_Float16*)carve((size_t)BLq * DTRq * 2);
    float*    dt32   = (float*)carve((size_t)BLq * DINq * 4);
    _Float16* y16    = (_Float16*)carve((size_t)BLq * DINq * 2);

    // 1) Weight conversions to f16 (L2-resident thereafter).
    {
        int n1 = 2 * DINq * Dq;
        f32_to_f16_kernel<<<(n1 + 255) / 256, 256, 0, stream>>>(in_proj_w, win16, n1, n1);
        int n2v = (DTRq + 2 * Nq) * DINq;          // 96 valid rows...
        int n2t = XDS * DINq;                      // ...padded to 128 rows
        f32_to_f16_kernel<<<(n2t + 255) / 256, 256, 0, stream>>>(x_proj_w, wxp16, n2v, n2t);
        int n3 = DINq * DTRq;
        f32_to_f16_kernel<<<(n3 + 255) / 256, 256, 0, stream>>>(dt_proj_w, wdt16, n3, n3);
        int n4 = Dq * DINq;
        f32_to_f16_kernel<<<(n4 + 255) / 256, 256, 0, stream>>>(out_proj_w, wout16, n4, n4);
    }

    // 2) resid = residual + hidden; h16 = LN(resid).
    add_ln_kernel<<<BLq, 256, 0, stream>>>(hidden, residual, norm_w, norm_b,
                                           out_resid, h16);

    // 3) xz = h @ in_proj_w^T     (8192 x 4096 x 1024)
    gemm_wmma_f16<<<dim3(2 * DINq / 64, BLq / 256), 256, 0, stream>>>(
        h16, win16, xz32, BLq, 2 * DINq, Dq);

    // 4) depthwise causal conv + SiLU on the x half of xz.
    conv_silu_kernel<<<(BLq * DINq) / 256, 256, 0, stream>>>(
        xz32, conv_w, conv_b, xc32, xc16);

    // 5) x_dbl = xconv @ x_proj_w^T   (8192 x 128 x 2048, N padded 96->128)
    gemm_wmma_f16<<<dim3(XDS / 64, BLq / 256), 256, 0, stream>>>(
        xc16, wxp16, xdbl32, BLq, XDS, DINq);

    // 6) dt_raw -> f16, then dt = softplus(dt_raw @ dt_proj_w^T + bias).
    extract_dt_kernel<<<(BLq * DTRq) / 256, 256, 0, stream>>>(xdbl32, dtr16);
    gemm_wmma_f16<<<dim3(DINq / 64, BLq / 256), 256, 0, stream>>>(
        dtr16, wdt16, dt32, BLq, DINq, DTRq);
    softplus_kernel<<<(BLq * DINq) / 256, 256, 0, stream>>>(dt32, dt_proj_b);

    // 7) Selective scan + fused (y + x*D)*silu(z); emits y16 + last_state.
    scan_kernel<<<Bq * (DINq / 256), 256, 0, stream>>>(
        dt32, xc32, xz32, xdbl32, A_log, D_param, y16, out_state);

    // 8) out = y @ out_proj_w^T    (8192 x 1024 x 2048)
    gemm_wmma_f16<<<dim3(Dq / 64, BLq / 256), 256, 0, stream>>>(
        y16, wout16, out_y, BLq, Dq, DINq);
}
